// VQGAN3D_torch_38594576122216
// MI455X (gfx1250) — matrix operand order repaired and checked by reading, not compiled, not run on hardware
//
#include <hip/hip_runtime.h>

typedef __attribute__((ext_vector_type(16))) __bf16 v16bf;
typedef __attribute__((ext_vector_type(8)))  float  v8f;

__device__ __forceinline__ int clampi(int v, int lo, int hi) {
    return v < lo ? lo : (v > hi ? hi : v);
}

union Frag { v16bf v; uint4 q[2]; };

// ---------------------------------------------------------------------------
// fp32 -> bf16 elementwise convert
// ---------------------------------------------------------------------------
__global__ void cvt_bf16_kernel(const float* __restrict__ src, __bf16* __restrict__ dst, size_t n) {
    size_t i = (size_t)blockIdx.x * blockDim.x + threadIdx.x;
    size_t stride = (size_t)gridDim.x * blockDim.x;
    for (; i < n; i += stride) dst[i] = (__bf16)src[i];
}

// ---------------------------------------------------------------------------
// fp32 [Cout][Cin][K3] -> bf16 [Cout][K3][Cin] (tap-major K) repack
// ---------------------------------------------------------------------------
__global__ void repack_w_kernel(const float* __restrict__ src, __bf16* __restrict__ dst,
                                int Cin, int K3, int total) {
    int i = blockIdx.x * blockDim.x + threadIdx.x;
    int stride = gridDim.x * blockDim.x;
    for (; i < total; i += stride) {
        int co = i / (Cin * K3);
        int r  = i - co * (Cin * K3);
        int ci = r / K3;
        int t  = r - ci * K3;
        dst[(size_t)co * (Cin * K3) + t * Cin + ci] = (__bf16)src[i];
    }
}

// ---------------------------------------------------------------------------
// GroupNorm(32 groups) + SiLU, fp32 in -> bf16 out. grid = N*32 blocks, 256 thr
// ---------------------------------------------------------------------------
__global__ __launch_bounds__(256) void gn_silu_kernel(
    const float* __restrict__ x, const float* __restrict__ gamma,
    const float* __restrict__ beta, __bf16* __restrict__ out,
    int C, int DHW)
{
    int n   = blockIdx.x >> 5;
    int g   = blockIdx.x & 31;
    int cpg = C >> 5;
    size_t base = ((size_t)n * C + (size_t)g * cpg) * (size_t)DHW;
    int cnt = cpg * DHW;

    float s = 0.f, ss = 0.f;
    for (int i = threadIdx.x; i < cnt; i += 256) {
        float v = x[base + i];
        s += v; ss += v * v;
    }
    __shared__ float rs[256], rq[256];
    rs[threadIdx.x] = s; rq[threadIdx.x] = ss;
    __syncthreads();
    for (int off = 128; off > 0; off >>= 1) {
        if (threadIdx.x < off) {
            rs[threadIdx.x] += rs[threadIdx.x + off];
            rq[threadIdx.x] += rq[threadIdx.x + off];
        }
        __syncthreads();
    }
    float mean = rs[0] / (float)cnt;
    float var  = rq[0] / (float)cnt - mean * mean;
    float inv  = rsqrtf(var + 1e-6f);

    for (int c = 0; c < cpg; ++c) {
        float ga = gamma[g * cpg + c], be = beta[g * cpg + c];
        size_t cb = base + (size_t)c * DHW;
        for (int i = threadIdx.x; i < DHW; i += 256) {
            float v = (x[cb + i] - mean) * inv * ga + be;
            v = v / (1.f + __expf(-v));          // SiLU
            out[cb + i] = (__bf16)v;
        }
    }
}

// ---------------------------------------------------------------------------
// Implicit-GEMM conv3d with WMMA bf16->f32, tap-major K (k = tap*Cin + ci).
//  Block = 64 c_out x 64 positions; wave owns 16 c_out rows, 4 wmma / K chunk.
//  FAST (Cin%16==0): a 16-element half-chunk lies inside one tap -> clamped
//  coords computed once per chunk, loads are base + j*DHWi increments.
//  Software-pipelined: next chunk's global loads issued before current WMMAs.
// ---------------------------------------------------------------------------
template <int KS, bool FAST>
__global__ __launch_bounds__(128) void conv3d_wmma_kernel(
    const __bf16* __restrict__ in, const __bf16* __restrict__ w,
    const float* __restrict__ bias, const float* __restrict__ res,
    float* __restrict__ out,
    int Cin, int Cout, int Di, int Hi, int Wi,
    int Do, int Ho, int Wo, int stride, int padlo, int P, int cinShift)
{
    constexpr int K3 = KS * KS * KS;
    __shared__ __align__(16) __bf16 Aw[64][40];   // [c_out][K]
    __shared__ __align__(16) __bf16 Bt[64][40];   // [position][K] (transposed)

    const int tid  = threadIdx.x;
    const int lane = tid & 31;
    const int wave = tid >> 5;
    const int hf   = lane >> 4;
    const int p_base  = blockIdx.x * 64;
    const int co_base = blockIdx.y * 64;
    const int Ktot = Cin * K3;
    const int cinMask = (1 << cinShift) - 1;
    const size_t DHWi = (size_t)Di * Hi * Wi;

    const unsigned short* in16 = (const unsigned short*)in;
    const unsigned short* w16  = (const unsigned short*)w;

    // ---- B-gather mapping: pos = tid&63, K half = tid>>6 ----
    const int pos   = tid & 63;
    const int khalf = tid >> 6;
    int pg = p_base + pos;
    int pc = pg < P ? pg : (P - 1);                 // clamp -> always valid addr
    const int HW = Ho * Wo;
    int n  = pc / (Do * HW); int rem = pc % (Do * HW);
    int od = rem / HW; rem %= HW;
    int oh = rem / Wo; int ow = rem % Wo;
    int idb = od * stride - padlo;
    int ihb = oh * stride - padlo;
    int iwb = ow * stride - padlo;
    const size_t nbase = (size_t)n * Cin * DHWi;

    // ---- A-load mapping: row = tid>>1, 16 k at (tid&1)*16 ----
    const int am  = tid >> 1;
    const int alk = (tid & 1) * 16;
    const int aco = co_base + am;
    const size_t arow_base = (size_t)(aco < Cout ? aco : Cout - 1) * Ktot;

    auto fetchA = [&](int kc, unsigned short* tmp) {
        int kk0 = kc + alk;
#pragma unroll
        for (int j = 0; j < 16; ++j) {
            int kk = kk0 + j;
            int ks = kk < Ktot ? kk : Ktot - 1;
            unsigned short raw = w16[arow_base + ks];      // unconditional load
            tmp[j] = kk < Ktot ? raw : (unsigned short)0;  // K-tail zero (cndmask)
        }
    };
    auto fetchB = [&](int kc, unsigned short* tmp) {
        int kk0 = kc + khalf * 16;
        if (FAST) {
            // whole half-chunk shares one tap: coords once, stride-DHWi loads
            int t   = kk0 >> cinShift;
            int ci0 = kk0 & cinMask;
            int kd = t / (KS * KS);
            int kh = (t / KS) % KS;
            int kw = t % KS;
            int id = clampi(idb + kd, 0, Di - 1);
            int ih = clampi(ihb + kh, 0, Hi - 1);
            int iw = clampi(iwb + kw, 0, Wi - 1);
            size_t base = nbase + (size_t)ci0 * DHWi + ((size_t)id * Hi + ih) * Wi + iw;
#pragma unroll
            for (int j = 0; j < 16; ++j)
                tmp[j] = in16[base + (size_t)j * DHWi];
        } else {
#pragma unroll
            for (int j = 0; j < 16; ++j) {
                int kk = kk0 + j;
                int ks = kk < Ktot ? kk : Ktot - 1;
                int t  = ks >> cinShift;
                int ci = ks & cinMask;
                int kd = t / (KS * KS);
                int kh = (t / KS) % KS;
                int kw = t % KS;
                int id = clampi(idb + kd, 0, Di - 1);
                int ih = clampi(ihb + kh, 0, Hi - 1);
                int iw = clampi(iwb + kw, 0, Wi - 1);
                tmp[j] = in16[nbase + (size_t)ci * DHWi + ((size_t)id * Hi + ih) * Wi + iw];
            }
        }
    };

    v8f zero = {0.f, 0.f, 0.f, 0.f, 0.f, 0.f, 0.f, 0.f};
    v8f accs[4];
#pragma unroll
    for (int g = 0; g < 4; ++g) accs[g] = zero;

    __align__(16) unsigned short ta[16], tb[16];
    fetchA(0, ta);
    fetchB(0, tb);

    for (int kc = 0; kc < Ktot; kc += 32) {
        // ---- commit staged registers to LDS ----
        {
            uint4* dA = reinterpret_cast<uint4*>(&Aw[am][alk]);
            dA[0] = *reinterpret_cast<const uint4*>(&ta[0]);
            dA[1] = *reinterpret_cast<const uint4*>(&ta[8]);
            uint4* dB = reinterpret_cast<uint4*>(&Bt[pos][khalf * 16]);
            dB[0] = *reinterpret_cast<const uint4*>(&tb[0]);
            dB[1] = *reinterpret_cast<const uint4*>(&tb[8]);
        }
        __syncthreads();

        // ---- issue next chunk's global loads (overlap with WMMA below) ----
        if (kc + 32 < Ktot) {
            fetchA(kc + 32, ta);
            fetchB(kc + 32, tb);
        }

        // ---- one A fragment, 4 B fragments, 4 wmma ----
        Frag a;
        {
            const __bf16* arow = &Aw[wave * 16 + (lane & 15)][0];
            a.q[0] = *reinterpret_cast<const uint4*>(arow + hf * 16);
            a.q[1] = *reinterpret_cast<const uint4*>(arow + hf * 16 + 8);
        }
        const int bcol = lane & 15;
#pragma unroll
        for (int g = 0; g < 4; ++g) {
            Frag b;
            const __bf16* brow = &Bt[g * 16 + bcol][0];
            b.q[0] = *reinterpret_cast<const uint4*>(brow + hf * 16);
            b.q[1] = *reinterpret_cast<const uint4*>(brow + hf * 16 + 8);
            accs[g] = __builtin_amdgcn_wmma_f32_16x16x32_bf16(
                false, a.v, false, b.v, (short)0, accs[g], false, false);
        }
        __syncthreads();
    }

    // ---- epilogue: D row = vgpr + 8*hf, col = lane&15 ----
#pragma unroll
    for (int g = 0; g < 4; ++g) {
        int colp = p_base + g * 16 + (lane & 15);
        if (colp < P) {
            int n2  = colp / (Do * HW); int r2 = colp % (Do * HW);
            int od2 = r2 / HW; r2 %= HW;
            int oh2 = r2 / Wo; int ow2 = r2 % Wo;
#pragma unroll
            for (int r = 0; r < 8; ++r) {
                int co = co_base + wave * 16 + r + hf * 8;
                if (co < Cout) {
                    size_t oidx = ((((size_t)n2 * Cout + co) * Do + od2) * Ho + oh2) * Wo + ow2;
                    float v = accs[g][r] + bias[co];
                    if (res) v += res[oidx];
                    out[oidx] = v;
                }
            }
        }
    }
}

// ---------------------------------------------------------------------------
// Transposed conv3d (k=4, s=2, torch padding=3 after replicate-pad(1)) as
// 8 parity sub-convolutions: 2 taps/dim -> K = tap*Cin + ci (tap-major).
// Cin is always 128 here -> half-chunks never straddle taps: per-chunk
// coordinate math, stride-DHWi B loads, stride-(Cout*64) A loads (no repack).
// Weights layout [Cin][Cout][4][4][4], flipped tap = 3 - parity - 2*t.
// ---------------------------------------------------------------------------
__global__ __launch_bounds__(128) void convt3d_wmma_kernel(
    const __bf16* __restrict__ in, const __bf16* __restrict__ w,
    const float* __restrict__ bias, float* __restrict__ out,
    int Cin, int Cout, int Di, int Hi, int Wi, int P, int cinShift)
{
    __shared__ __align__(16) __bf16 Aw[64][40];
    __shared__ __align__(16) __bf16 Bt[64][40];

    const int tid  = threadIdx.x;
    const int lane = tid & 31;
    const int wave = tid >> 5;
    const int hf   = lane >> 4;
    const int p_base  = blockIdx.x * 64;
    const int co_base = blockIdx.y * 64;
    const int pd = (blockIdx.z >> 2) & 1;
    const int ph = (blockIdx.z >> 1) & 1;
    const int pw = blockIdx.z & 1;
    const int Ktot = Cin * 8;
    const int cinMask = (1 << cinShift) - 1;
    const size_t DHWi = (size_t)Di * Hi * Wi;
    const int Do = 2 * Di, Ho = 2 * Hi, Wo = 2 * Wi;

    const unsigned short* in16 = (const unsigned short*)in;
    const unsigned short* w16  = (const unsigned short*)w;

    const int pos   = tid & 63;
    const int khalf = tid >> 6;
    int pg = p_base + pos;
    int pc = pg < P ? pg : (P - 1);
    const int HW = Hi * Wi;
    int n   = pc / (Di * HW); int rem = pc % (Di * HW);
    int odp = rem / HW; rem %= HW;
    int ohp = rem / Wi; int owp = rem % Wi;
    const size_t nbase = (size_t)n * Cin * DHWi;

    const int am  = tid >> 1;
    const int alk = (tid & 1) * 16;
    const int aco = co_base + am;
    const int acs = aco < Cout ? aco : Cout - 1;

    auto fetchA = [&](int kc, unsigned short* tmp) {
        int kk0 = kc + alk;
        int t   = kk0 >> cinShift;
        int ci0 = kk0 & cinMask;
        int td = (t >> 2) & 1, th = (t >> 1) & 1, tw = t & 1;
        int kd = 3 - pd - 2 * td;
        int kh = 3 - ph - 2 * th;
        int kw = 3 - pw - 2 * tw;
        size_t base = ((size_t)ci0 * Cout + acs) * 64 + kd * 16 + kh * 4 + kw;
        const size_t strideA = (size_t)Cout * 64;
#pragma unroll
        for (int j = 0; j < 16; ++j)
            tmp[j] = w16[base + (size_t)j * strideA];
    };
    auto fetchB = [&](int kc, unsigned short* tmp) {
        int kk0 = kc + khalf * 16;
        int t   = kk0 >> cinShift;
        int ci0 = kk0 & cinMask;
        int td = (t >> 2) & 1, th = (t >> 1) & 1, tw = t & 1;
        int id = clampi(odp + pd + td - 1, 0, Di - 1);
        int ih = clampi(ohp + ph + th - 1, 0, Hi - 1);
        int iw = clampi(owp + pw + tw - 1, 0, Wi - 1);
        size_t base = nbase + (size_t)ci0 * DHWi + ((size_t)id * Hi + ih) * Wi + iw;
#pragma unroll
        for (int j = 0; j < 16; ++j)
            tmp[j] = in16[base + (size_t)j * DHWi];
    };

    v8f zero = {0.f, 0.f, 0.f, 0.f, 0.f, 0.f, 0.f, 0.f};
    v8f accs[4];
#pragma unroll
    for (int g = 0; g < 4; ++g) accs[g] = zero;

    __align__(16) unsigned short ta[16], tb[16];
    fetchA(0, ta);
    fetchB(0, tb);

    for (int kc = 0; kc < Ktot; kc += 32) {
        {
            uint4* dA = reinterpret_cast<uint4*>(&Aw[am][alk]);
            dA[0] = *reinterpret_cast<const uint4*>(&ta[0]);
            dA[1] = *reinterpret_cast<const uint4*>(&ta[8]);
            uint4* dB = reinterpret_cast<uint4*>(&Bt[pos][khalf * 16]);
            dB[0] = *reinterpret_cast<const uint4*>(&tb[0]);
            dB[1] = *reinterpret_cast<const uint4*>(&tb[8]);
        }
        __syncthreads();

        if (kc + 32 < Ktot) {
            fetchA(kc + 32, ta);
            fetchB(kc + 32, tb);
        }

        Frag a;
        {
            const __bf16* arow = &Aw[wave * 16 + (lane & 15)][0];
            a.q[0] = *reinterpret_cast<const uint4*>(arow + hf * 16);
            a.q[1] = *reinterpret_cast<const uint4*>(arow + hf * 16 + 8);
        }
        const int bcol = lane & 15;
#pragma unroll
        for (int g = 0; g < 4; ++g) {
            Frag b;
            const __bf16* brow = &Bt[g * 16 + bcol][0];
            b.q[0] = *reinterpret_cast<const uint4*>(brow + hf * 16);
            b.q[1] = *reinterpret_cast<const uint4*>(brow + hf * 16 + 8);
            accs[g] = __builtin_amdgcn_wmma_f32_16x16x32_bf16(
                false, a.v, false, b.v, (short)0, accs[g], false, false);
        }
        __syncthreads();
    }

#pragma unroll
    for (int g = 0; g < 4; ++g) {
        int colp = p_base + g * 16 + (lane & 15);
        if (colp < P) {
            int n2   = colp / (Di * HW); int r2 = colp % (Di * HW);
            int odp2 = r2 / HW; r2 %= HW;
            int ohp2 = r2 / Wi; int owp2 = r2 % Wi;
            int od = 2 * odp2 + pd, oh = 2 * ohp2 + ph, ow = 2 * owp2 + pw;
#pragma unroll
            for (int r = 0; r < 8; ++r) {
                int co = co_base + wave * 16 + r + hf * 8;
                if (co < Cout) {
                    size_t oidx = ((((size_t)n2 * Cout + co) * Do + od) * Ho + oh) * Wo + ow;
                    out[oidx] = accs[g][r] + bias[co];
                }
            }
        }
    }
}

// ---------------------------------------------------------------------------
// VQ: nearest-codebook lookup; forward value == codebook[argmin].
// Codebook (2048 x 8 fp32 = 64 KB) staged in LDS (320 KB/WGP on CDNA5).
// ---------------------------------------------------------------------------
__global__ __launch_bounds__(256) void vq_kernel(
    const float* __restrict__ z, const float* __restrict__ cb,
    float* __restrict__ zq, int P, int dhw)
{
    __shared__ float lcb[2048 * 8];
    for (int i = threadIdx.x; i < 2048 * 8; i += 256) lcb[i] = cb[i];
    __syncthreads();

    int p = blockIdx.x * 256 + threadIdx.x;
    if (p >= P) return;
    int n = p / dhw, s = p % dhw;

    float zv[8];
#pragma unroll
    for (int e = 0; e < 8; ++e)
        zv[e] = z[((size_t)(n * 8 + e)) * dhw + s];

    float best = 3.4e38f;
    int bi = 0;
    for (int c = 0; c < 2048; ++c) {
        float d = 0.f;
#pragma unroll
        for (int e = 0; e < 8; ++e) {
            float t = zv[e] - lcb[c * 8 + e];
            d += t * t;
        }
        if (d < best) { best = d; bi = c; }   // first-min tie-break like argmin
    }
#pragma unroll
    for (int e = 0; e < 8; ++e)
        zq[((size_t)(n * 8 + e)) * dhw + s] = lcb[bi * 8 + e];
}

// ---------------------------------------------------------------------------
// Host orchestration
// ---------------------------------------------------------------------------
extern "C" void kernel_launch(void* const* d_in, const int* in_sizes, int n_in,
                              void* d_out, int out_size, void* d_ws, size_t ws_size,
                              hipStream_t stream)
{
    (void)in_sizes; (void)n_in; (void)out_size; (void)ws_size;

    const float* X = (const float*)d_in[0];
    auto F = [&](int i) { return (const float*)d_in[i]; };

    char* ws = (char*)d_ws;
    const size_t SF = (size_t)2 * 64 * 32 * 96 * 96 * sizeof(float); // 151 MB
    float*  F0 = (float*)(ws + 0 * SF);
    float*  F1 = (float*)(ws + 1 * SF);
    float*  F2 = (float*)(ws + 2 * SF);
    __bf16* G  = (__bf16*)(ws + 3 * SF);              // SF/2 bytes of bf16
    __bf16* WB = (__bf16*)(ws + 3 * SF + SF / 2);     // bf16 weights (~11 MB)

    // --- stage all weights to bf16; standard convs repacked tap-major ---
    // spec: {input idx, Cout, Cin, K3}; K3==0 -> plain convert (convT weights)
    struct WS { int idx; int Cout, Cin, K3; };
    const WS wl[20] = {
        {1, 32, 1, 27},    {3, 64, 32, 64},   {7, 64, 64, 27},  {11, 64, 64, 27},
        {13, 128, 64, 64}, {17, 128, 128, 27},{21, 128, 128, 27},
        {25, 8, 128, 1},   {28, 128, 8, 1},   {32, 0, 0, 0},    // du0: convT
        {36, 128, 128, 27},{40, 128, 128, 27},{44, 128, 128, 27},{48, 128, 128, 27},
        {50, 0, 0, 0},                                           // du1: convT
        {54, 64, 64, 27},  {58, 64, 64, 27},  {62, 64, 64, 27}, {66, 64, 64, 27},
        {68, 1, 64, 27}
    };
    const size_t wn[20] = {
        864, 131072, 110592, 110592, 524288, 442368, 442368, 1024, 1024, 1048576,
        442368, 442368, 442368, 442368, 524288, 110592, 110592, 110592, 110592, 1728
    };
    __bf16* wp[20];
    {
        size_t off = 0;
        for (int i = 0; i < 20; ++i) {
            wp[i] = WB + off;
            if (wl[i].K3 == 0) {
                cvt_bf16_kernel<<<128, 256, 0, stream>>>(F(wl[i].idx), wp[i], wn[i]);
            } else {
                repack_w_kernel<<<128, 256, 0, stream>>>(F(wl[i].idx), wp[i],
                    wl[i].Cin, wl[i].K3, (int)wn[i]);
            }
            off += wn[i];
        }
    }
    __bf16 *Wef = wp[0], *Wed0 = wp[1], *Wer0c1 = wp[2], *Wer0c2 = wp[3], *Wed1 = wp[4],
           *Wer1c1 = wp[5], *Wer1c2 = wp[6], *Wpre = wp[7], *Wpost = wp[8], *Wdu0 = wp[9],
           *Wdr0a1 = wp[10], *Wdr0a2 = wp[11], *Wdr0b1 = wp[12], *Wdr0b2 = wp[13],
           *Wdu1 = wp[14], *Wdr1a1 = wp[15], *Wdr1a2 = wp[16], *Wdr1b1 = wp[17],
           *Wdr1b2 = wp[18], *Wlast = wp[19];

    auto log2i = [](int v) { int s = 0; while ((1 << s) < v) ++s; return s; };

    auto conv = [&](int KS, const __bf16* in, const __bf16* w, const float* bias,
                    const float* res, float* out, int Cin, int Cout,
                    int Di, int Hi, int Wi, int Do, int Ho, int Wo,
                    int stride, int padlo) {
        int P = 2 * Do * Ho * Wo;
        int cs = log2i(Cin);
        bool fast = (Cin % 16) == 0;
        dim3 grid((P + 63) / 64, (Cout + 63) / 64);
        if (KS == 3) {
            if (fast)
                conv3d_wmma_kernel<3, true><<<grid, 128, 0, stream>>>(in, w, bias, res, out,
                    Cin, Cout, Di, Hi, Wi, Do, Ho, Wo, stride, padlo, P, cs);
            else
                conv3d_wmma_kernel<3, false><<<grid, 128, 0, stream>>>(in, w, bias, res, out,
                    Cin, Cout, Di, Hi, Wi, Do, Ho, Wo, stride, padlo, P, cs);
        } else if (KS == 4) {
            conv3d_wmma_kernel<4, true><<<grid, 128, 0, stream>>>(in, w, bias, res, out,
                Cin, Cout, Di, Hi, Wi, Do, Ho, Wo, stride, padlo, P, cs);
        } else {
            if (fast)
                conv3d_wmma_kernel<1, true><<<grid, 128, 0, stream>>>(in, w, bias, res, out,
                    Cin, Cout, Di, Hi, Wi, Do, Ho, Wo, stride, padlo, P, cs);
            else
                conv3d_wmma_kernel<1, false><<<grid, 128, 0, stream>>>(in, w, bias, res, out,
                    Cin, Cout, Di, Hi, Wi, Do, Ho, Wo, stride, padlo, P, cs);
        }
    };
    auto convt = [&](const __bf16* in, const __bf16* w, const float* bias, float* out,
                     int Cin, int Cout, int Di, int Hi, int Wi) {
        int P = 2 * Di * Hi * Wi;
        dim3 grid((P + 63) / 64, (Cout + 63) / 64, 8);
        convt3d_wmma_kernel<<<grid, 128, 0, stream>>>(in, w, bias, out, Cin, Cout,
                                                      Di, Hi, Wi, P, log2i(Cin));
    };
    auto gns = [&](const float* x, const float* g, const float* b, __bf16* o,
                   int C, int DHW) {
        gn_silu_kernel<<<64, 256, 0, stream>>>(x, g, b, o, C, DHW);
    };
    auto cvt = [&](const float* s, __bf16* d, size_t n) {
        cvt_bf16_kernel<<<512, 256, 0, stream>>>(s, d, n);
    };

    // ---------------- Encoder ----------------
    cvt(X, G, (size_t)2 * 1 * 294912);
    conv(3, G, Wef, F(2), nullptr, F0, 1, 32, 32, 96, 96, 32, 96, 96, 1, 1);
    cvt(F0, G, (size_t)2 * 32 * 294912);
    conv(4, G, Wed0, F(4), nullptr, F1, 32, 64, 32, 96, 96, 16, 48, 48, 2, 1);
    // er0
    gns(F1, F(5), F(6), G, 64, 36864);
    conv(3, G, Wer0c1, F(8), nullptr, F0, 64, 64, 16, 48, 48, 16, 48, 48, 1, 1);
    gns(F0, F(9), F(10), G, 64, 36864);
    conv(3, G, Wer0c2, F(12), F1, F2, 64, 64, 16, 48, 48, 16, 48, 48, 1, 1);
    cvt(F2, G, (size_t)2 * 64 * 36864);
    conv(4, G, Wed1, F(14), nullptr, F0, 64, 128, 16, 48, 48, 8, 24, 24, 2, 1);
    // er1
    gns(F0, F(15), F(16), G, 128, 4608);
    conv(3, G, Wer1c1, F(18), nullptr, F1, 128, 128, 8, 24, 24, 8, 24, 24, 1, 1);
    gns(F1, F(19), F(20), G, 128, 4608);
    conv(3, G, Wer1c2, F(22), F0, F2, 128, 128, 8, 24, 24, 8, 24, 24, 1, 1);
    // efin + pre (1x1x1)
    gns(F2, F(23), F(24), G, 128, 4608);
    conv(1, G, Wpre, F(26), nullptr, F0, 128, 8, 8, 24, 24, 8, 24, 24, 1, 0);
    // VQ bottleneck
    vq_kernel<<<(9216 + 255) / 256, 256, 0, stream>>>(F0, F(27), F1, 9216, 4608);
    // post (1x1x1)
    cvt(F1, G, (size_t)2 * 8 * 4608);
    conv(1, G, Wpost, F(29), nullptr, F0, 8, 128, 8, 24, 24, 8, 24, 24, 1, 0);

    // ---------------- Decoder ----------------
    gns(F0, F(30), F(31), G, 128, 4608);
    convt(G, Wdu0, F(33), F1, 128, 128, 8, 24, 24);
    // dr0a
    gns(F1, F(34), F(35), G, 128, 36864);
    conv(3, G, Wdr0a1, F(37), nullptr, F0, 128, 128, 16, 48, 48, 16, 48, 48, 1, 1);
    gns(F0, F(38), F(39), G, 128, 36864);
    conv(3, G, Wdr0a2, F(41), F1, F2, 128, 128, 16, 48, 48, 16, 48, 48, 1, 1);
    // dr0b
    gns(F2, F(42), F(43), G, 128, 36864);
    conv(3, G, Wdr0b1, F(45), nullptr, F0, 128, 128, 16, 48, 48, 16, 48, 48, 1, 1);
    gns(F0, F(46), F(47), G, 128, 36864);
    conv(3, G, Wdr0b2, F(49), F2, F1, 128, 128, 16, 48, 48, 16, 48, 48, 1, 1);
    // du1
    cvt(F1, G, (size_t)2 * 128 * 36864);
    convt(G, Wdu1, F(51), F0, 128, 64, 16, 48, 48);
    // dr1a
    gns(F0, F(52), F(53), G, 64, 294912);
    conv(3, G, Wdr1a1, F(55), nullptr, F1, 64, 64, 32, 96, 96, 32, 96, 96, 1, 1);
    gns(F1, F(56), F(57), G, 64, 294912);
    conv(3, G, Wdr1a2, F(59), F0, F2, 64, 64, 32, 96, 96, 32, 96, 96, 1, 1);
    // dr1b
    gns(F2, F(60), F(61), G, 64, 294912);
    conv(3, G, Wdr1b1, F(63), nullptr, F1, 64, 64, 32, 96, 96, 32, 96, 96, 1, 1);
    gns(F1, F(64), F(65), G, 64, 294912);
    conv(3, G, Wdr1b2, F(67), F2, F0, 64, 64, 32, 96, 96, 32, 96, 96, 1, 1);
    // last conv -> d_out
    cvt(F0, G, (size_t)2 * 64 * 294912);
    conv(3, G, Wlast, F(69), nullptr, (float*)d_out, 64, 1, 32, 96, 96, 32, 96, 96, 1, 1);
}
